// IREmbeder_83305185673767
// MI455X (gfx1250) — matrix operand order repaired and checked by reading, not compiled, not run on hardware
//
#include <hip/hip_runtime.h>
#include <hip/hip_bf16.h>
#include <stdint.h>

typedef __attribute__((ext_vector_type(16))) __bf16 v16bf;
typedef __attribute__((ext_vector_type(8)))  float  v8f;

#define HDIM 256
#define NEG_INF_F (-1e30f)
#define WPK_DWORDS (16 * 8 * 32 * 8)   // 32768 dwords = 128 KB packed W

static __device__ inline unsigned short f2bfu(float f) {
    union { float f; uint32_t u; } x; x.f = f;
    return (unsigned short)(x.u >> 16);   // truncate to bf16
}

// Branchless tanh: 1 - 2/(e^{2x}+1).  Saturates to +/-1, propagates NaN,
// lowers to v_exp_f32 + rcp instead of libm's divergent branch tree.
static __device__ inline float fast_tanh(float x) {
    float e = __expf(2.0f * x);
    return 1.0f - 2.0f / (e + 1.0f);
}

// ---------------------------------------------------------------------------
// Repack W[H,H] (f32, K rows x N cols) into bf16 pairs in this code's contiguous-k (k = 16h + s) WMMA
// B-matrix (32x16 KxN) per-lane VGPR layout:
// flat = (((nt*8)+kc)*32 + lane)*8 + v
//   lane 0-15 : N = nt*16 + lane,    K pair = kc*32 + 2v
//   lane16-31 : N = nt*16 + lane-16, K pair = kc*32 + 16 + 2v
// ---------------------------------------------------------------------------
__global__ void pack_w_kernel(const float* __restrict__ W, uint32_t* __restrict__ out) {
    int flat = blockIdx.x * blockDim.x + threadIdx.x;
    if (flat >= WPK_DWORDS) return;
    int v    = flat & 7;
    int lane = (flat >> 3) & 31;
    int kc   = (flat >> 8) & 7;
    int nt   = flat >> 11;
    int K = kc * 32 + 16 * (lane >> 4) + 2 * v;
    int n = nt * 16 + (lane & 15);
    unsigned short lo = f2bfu(W[(size_t)K * HDIM + n]);
    unsigned short hi = f2bfu(W[(size_t)(K + 1) * HDIM + n]);
    out[flat] = (uint32_t)lo | ((uint32_t)hi << 16);
}

// ---------------------------------------------------------------------------
// Fused attention pooling: scores = tanh(feat@W + b)·v + c, masked softmax,
// pooled = w^T feat, out = tanh(pooled).  One block per batch, 8 waves.
// Packed W staged global->LDS with async-LDS copies (ASYNCcnt), inner loop
// is ds_load_b128 + v_wmma_f32_16x16x32_bf16.
// ---------------------------------------------------------------------------
__global__ __launch_bounds__(256)
void attn_pool_kernel(const float* __restrict__ feat, int L,
                      const int* __restrict__ lens,
                      const uint32_t* __restrict__ Wpk,
                      const float* __restrict__ bvec,
                      const float* __restrict__ vvec,
                      const float* __restrict__ cptr,
                      float* __restrict__ out) {
    __shared__ uint32_t wlds[WPK_DWORDS];   // 128 KB of 320 KB WGP LDS
    __shared__ float sc[512];
    __shared__ float red[256];
    __shared__ float sbc[2];

    const int b    = blockIdx.x;
    const int tid  = threadIdx.x;
    const int lane = tid & 31;
    const int wid  = tid >> 5;
    const int half = lane >> 4;   // 0: lanes 0-15, 1: lanes 16-31
    const int lm   = lane & 15;
    const int len  = lens[b];
    const float cs = cptr[0];

    // Stage packed W into LDS with direct async global->LDS b128 copies:
    // no VGPR round-trip, tracked by ASYNCcnt.
    {
        #pragma unroll 4
        for (int i = tid; i < WPK_DWORDS / 4; i += 256) {
            uint32_t lds_addr =
                (uint32_t)(uintptr_t)(&wlds[(size_t)i * 4]);   // low 32b = LDS offset
            const uint4* gaddr = (const uint4*)Wpk + i;
            asm volatile("global_load_async_to_lds_b128 %0, %1, off"
                         :: "v"(lds_addr), "v"(gaddr) : "memory");
        }
        asm volatile("s_wait_asynccnt 0x0" ::: "memory");
    }
    __syncthreads();

    const float* fb = feat + (size_t)b * L * HDIM;
    const int ntiles = L >> 4;

    union UB { uint32_t u[8]; v16bf v; };

    for (int mt = wid; mt < ntiles; mt += 8) {
        const float* rowp = fb + (size_t)(mt * 16 + lm) * HDIM;

        // prefetch next row-tile's A panel row into caches while we compute
        if (mt + 8 < ntiles)
            __builtin_prefetch(fb + (size_t)((mt + 8) * 16 + lm) * HDIM, 0, 0);

        // A panel: 16 rows x 256 K in bf16, per-lane WMMA A layout.
        v16bf a[8];
        #pragma unroll
        for (int kc = 0; kc < 8; ++kc) {
            union { unsigned short s[16]; v16bf v; } ua;
            const int kb = kc * 32 + 16 * half;
            #pragma unroll
            for (int t = 0; t < 4; ++t) {
                float2 f0 = *(const float2*)(rowp + kb + 2 * t);
                ua.s[2 * t]     = f2bfu(f0.x);
                ua.s[2 * t + 1] = f2bfu(f0.y);
                float2 f1 = *(const float2*)(rowp + kb + 8 + 2 * t);
                ua.s[8 + 2 * t]     = f2bfu(f1.x);
                ua.s[8 + 2 * t + 1] = f2bfu(f1.y);
            }
            a[kc] = ua.v;
        }

        float partial[8];
        #pragma unroll
        for (int r = 0; r < 8; ++r) partial[r] = 0.f;

        for (int nt = 0; nt < 16; ++nt) {
            const int n    = nt * 16 + lm;
            const float bn = bvec[n];
            const float vn = vvec[n];

            // ping-pong B buffers so ds_load of kc+1 overlaps wmma of kc
            UB bb[2];
            {
                const uint32_t* bp = wlds + ((size_t)(nt << 3) * 32 + lane) * 8;
                uint4 q0 = *(const uint4*)bp;
                uint4 q1 = *(const uint4*)(bp + 4);
                bb[0].u[0] = q0.x; bb[0].u[1] = q0.y; bb[0].u[2] = q0.z; bb[0].u[3] = q0.w;
                bb[0].u[4] = q1.x; bb[0].u[5] = q1.y; bb[0].u[6] = q1.z; bb[0].u[7] = q1.w;
            }
            v8f c = {};
            #pragma unroll
            for (int kc = 0; kc < 8; ++kc) {
                if (kc < 7) {
                    const uint32_t* bp =
                        wlds + ((size_t)((nt << 3) + kc + 1) * 32 + lane) * 8;
                    uint4 q0 = *(const uint4*)bp;
                    uint4 q1 = *(const uint4*)(bp + 4);
                    UB& nx = bb[(kc + 1) & 1];
                    nx.u[0] = q0.x; nx.u[1] = q0.y; nx.u[2] = q0.z; nx.u[3] = q0.w;
                    nx.u[4] = q1.x; nx.u[5] = q1.y; nx.u[6] = q1.z; nx.u[7] = q1.w;
                }
                c = __builtin_amdgcn_wmma_f32_16x16x32_bf16(
                        false, a[kc], false, bb[kc & 1].v, (short)0, c, false, false);
            }
            // epilogue: tanh(s + b_n) * v_n accumulated per row (branchless)
            #pragma unroll
            for (int r = 0; r < 8; ++r)
                partial[r] += fast_tanh(c[r] + bn) * vn;
        }

        // reduce over the 16 N-lanes of each half (wave32 butterfly)
        #pragma unroll
        for (int m = 1; m <= 8; m <<= 1) {
            #pragma unroll
            for (int r = 0; r < 8; ++r)
                partial[r] += __shfl_xor(partial[r], m, 16);
        }
        if (lm == 0) {
            #pragma unroll
            for (int r = 0; r < 8; ++r)
                sc[mt * 16 + r + 8 * half] = partial[r] + cs;
        }
    }
    __syncthreads();

    // masked softmax over sc[0..L)
    float lmax = NEG_INF_F;
    for (int l = tid; l < L; l += 256)
        lmax = fmaxf(lmax, (l < len) ? sc[l] : NEG_INF_F);
    red[tid] = lmax;
    __syncthreads();
    for (int s = 128; s > 0; s >>= 1) {
        if (tid < s) red[tid] = fmaxf(red[tid], red[tid + s]);
        __syncthreads();
    }
    if (tid == 0) sbc[0] = red[0];
    __syncthreads();
    const float mx = sbc[0];

    float lsum = 0.f;
    for (int l = tid; l < L; l += 256)
        lsum += (l < len) ? __expf(sc[l] - mx) : 0.f;
    red[tid] = lsum;
    __syncthreads();
    for (int s = 128; s > 0; s >>= 1) {
        if (tid < s) red[tid] += red[tid + s];
        __syncthreads();
    }
    if (tid == 0) sbc[1] = red[0];
    __syncthreads();
    const float inv = 1.f / sbc[1];
    for (int l = tid; l < L; l += 256)
        sc[l] = (l < len) ? __expf(sc[l] - mx) * inv : 0.f;
    __syncthreads();

    // pooled[h] = sum_l w_l * feat[l,h]; thread tid owns h = tid (coalesced)
    float acc = 0.f;
    for (int l = 0; l < len; ++l)
        acc += sc[l] * fb[(size_t)l * HDIM + tid];
    out[(size_t)b * HDIM + tid] = fast_tanh(acc);
}

// ---------------------------------------------------------------------------
// Per-row contrastive loss: lse_i(logits) - logits_ii, logits = cos/0.05
// ---------------------------------------------------------------------------
__global__ __launch_bounds__(256)
void row_loss_kernel(const float* __restrict__ desc_repr,
                     const float* __restrict__ ir_repr,
                     float* __restrict__ row_loss) {
    __shared__ float dsh[256];
    __shared__ float red[256];
    __shared__ float diag[1];
    const int i = blockIdx.x;
    const int j = threadIdx.x;

    float dv = desc_repr[(size_t)i * 256 + j];
    dsh[j] = dv;
    red[j] = dv * dv;
    __syncthreads();
    for (int s = 128; s > 0; s >>= 1) {
        if (j < s) red[j] += red[j + s];
        __syncthreads();
    }
    const float nd = sqrtf(red[0]);
    __syncthreads();

    const float* irj = ir_repr + (size_t)j * 256;
    float dot = 0.f, n2 = 0.f;
    #pragma unroll 4
    for (int h = 0; h < 256; ++h) {
        float av = irj[h];
        dot += dsh[h] * av;
        n2  += av * av;
    }
    const float den   = fmaxf(nd * sqrtf(n2), 1e-8f);
    const float logit = (dot / den) * 20.0f;   // 1/0.05
    if (j == i) diag[0] = logit;

    red[j] = logit;
    __syncthreads();
    for (int s = 128; s > 0; s >>= 1) {
        if (j < s) red[j] = fmaxf(red[j], red[j + s]);
        __syncthreads();
    }
    const float mx = red[0];
    __syncthreads();
    red[j] = __expf(logit - mx);
    __syncthreads();
    for (int s = 128; s > 0; s >>= 1) {
        if (j < s) red[j] += red[j + s];
        __syncthreads();
    }
    if (j == 0) row_loss[i] = logf(red[0]) + mx - diag[0];
}

__global__ void final_loss_kernel(const float* __restrict__ row_loss,
                                  float* __restrict__ out) {
    __shared__ float red[256];
    const int t = threadIdx.x;
    red[t] = row_loss[t];
    __syncthreads();
    for (int s = 128; s > 0; s >>= 1) {
        if (t < s) red[t] += red[t + s];
        __syncthreads();
    }
    if (t == 0) out[0] = red[0] * (1.0f / 256.0f);
}

// ---------------------------------------------------------------------------
extern "C" void kernel_launch(void* const* d_in, const int* in_sizes, int n_in,
                              void* d_out, int out_size, void* d_ws, size_t ws_size,
                              hipStream_t stream) {
    (void)in_sizes; (void)n_in; (void)out_size; (void)ws_size;

    const float* code_feat = (const float*)d_in[0];   // [256,512,256]
    const float* desc_feat = (const float*)d_in[1];   // [256,128,256]
    const int*   ir_lens   = (const int*)d_in[2];
    const int*   de_lens   = (const int*)d_in[3];
    const float* W1 = (const float*)d_in[4];
    const float* b1 = (const float*)d_in[5];
    const float* v1 = (const float*)d_in[6];
    const float* c1 = (const float*)d_in[7];
    const float* W2 = (const float*)d_in[8];
    const float* b2 = (const float*)d_in[9];
    const float* v2 = (const float*)d_in[10];
    const float* c2 = (const float*)d_in[11];

    uint8_t* ws = (uint8_t*)d_ws;
    uint32_t* Wpk1      = (uint32_t*)ws;               // 128 KB
    uint32_t* Wpk2      = (uint32_t*)(ws + 131072);    // 128 KB
    float*    ir_repr   = (float*)(ws + 262144);       // 256 KB
    float*    desc_repr = (float*)(ws + 524288);       // 256 KB
    float*    row_loss  = (float*)(ws + 786432);       // 1 KB

    pack_w_kernel<<<WPK_DWORDS / 256, 256, 0, stream>>>(W1, Wpk1);
    pack_w_kernel<<<WPK_DWORDS / 256, 256, 0, stream>>>(W2, Wpk2);

    attn_pool_kernel<<<256, 256, 0, stream>>>(code_feat, 512, ir_lens, Wpk1, b1, v1, c1, ir_repr);
    attn_pool_kernel<<<256, 256, 0, stream>>>(desc_feat, 128, de_lens, Wpk2, b2, v2, c2, desc_repr);

    row_loss_kernel<<<256, 256, 0, stream>>>(desc_repr, ir_repr, row_loss);
    final_loss_kernel<<<1, 256, 0, stream>>>(row_loss, (float*)d_out);
}
